// RoPEAttention3D_8735963480744
// MI455X (gfx1250) — matrix operand order repaired and checked by reading, not compiled, not run on hardware
//
#include <hip/hip_runtime.h>
#include <stdint.h>

#define CDIM 256
#define NTOK 4096
#define NHEAD 4
#define HD 64

typedef __attribute__((ext_vector_type(16))) _Float16 v16h;
typedef __attribute__((ext_vector_type(8)))  _Float16 v8h;
typedef __attribute__((ext_vector_type(8)))  float    v8f;
typedef __attribute__((ext_vector_type(4)))  float    v4f;

union V16H { v16h v; v4f f[2]; };
union V8H  { v8h v; _Float16 e[8]; };

__device__ __forceinline__ v8f zf8() {
  v8f z = {0.f,0.f,0.f,0.f,0.f,0.f,0.f,0.f};
  return z;
}

// A-operand (16x32 f16, M x K), row-major buffer, contiguous K, leading dim ld.
// lane<16: row=lane, halves 0..7=K0..7, 8..15=K16..23; lane>=16: row=lane-16, K8..15 / K24..31.
__device__ __forceinline__ v16h load_a16(const _Float16* p0, int ld) {
  const int lane = threadIdx.x & 31;
  const int row  = lane & 15;
  const _Float16* p = p0 + (size_t)row * ld + ((lane & 16) ? 8 : 0);
  V16H u;
  u.f[0] = *(const v4f*)(p);
  u.f[1] = *(const v4f*)(p + 16);
  return u.v;
}

// B-operand (32x16 f16, K x N) from transposed matrix BT[n][k] row-major (contiguous K).
// lane<16: col=lane, K0..7 / K16..23; lane>=16: col=lane-16, K8..15 / K24..31.
__device__ __forceinline__ v16h load_bt16(const _Float16* p0, int ld) {
  const int lane = threadIdx.x & 31;
  const int col  = lane & 15;
  const _Float16* p = p0 + (size_t)col * ld + ((lane & 16) ? 8 : 0);
  V16H u;
  u.f[0] = *(const v4f*)(p);
  u.f[1] = *(const v4f*)(p + 16);
  return u.v;
}

__device__ __forceinline__ v8f wmma16(v16h a, v16h b, v8f c) {
  return __builtin_amdgcn_wmma_f32_16x16x32_f16(false, a, false, b, (short)0, c, false, false);
}

// --- CDNA5 async global->LDS copy (ASYNCcnt) ---------------------------------
// LDS byte address = low 32 bits of the generic pointer (ISA 10.2: LDS aperture
// keeps the LDS offset in addr[31:0]).
__device__ __forceinline__ uint32_t lds_addr32(const void* p) {
  return (uint32_t)(uintptr_t)p;
}
__device__ __forceinline__ void async_ld_b128(const void* lds_dst, const void* gsrc) {
  asm volatile("global_load_async_to_lds_b128 %0, %1, off"
               :: "v"(lds_addr32(lds_dst)), "v"((uint64_t)(uintptr_t)gsrc)
               : "memory");
}
__device__ __forceinline__ void wait_async0() {
  asm volatile("s_wait_asynccnt 0x0" ::: "memory");
}

// ---------------- kernel 0: convert the 4 weight matrices to f16 -------------
__global__ __launch_bounds__(256) void cvt_w_kernel(
    const float* __restrict__ Wq, const float* __restrict__ Wk,
    const float* __restrict__ Wv, const float* __restrict__ Wo,
    _Float16* __restrict__ dst) {
  int i = blockIdx.x * 256 + threadIdx.x;            // 4*65536 total
  int w = i >> 16, off = i & 65535;
  const float* s = (w == 0) ? Wq : (w == 1) ? Wk : (w == 2) ? Wv : Wo;
  dst[i] = (_Float16)s[off];
}

// ---------------- kernel 1: x (B,C,N) f32 -> xT (B,N,C) f16 ------------------
__global__ __launch_bounds__(256) void xpose_kernel(
    const float* __restrict__ x, _Float16* __restrict__ xT) {
  __shared__ float tile[16][17];
  int b  = blockIdx.z;
  int c0 = blockIdx.y * 16;
  int n0 = blockIdx.x * 16;
  int tx = threadIdx.x & 15, ty = threadIdx.x >> 4;
  tile[ty][tx] = x[((size_t)b * CDIM + c0 + ty) * NTOK + n0 + tx];
  __syncthreads();
  xT[((size_t)b * NTOK + n0 + ty) * CDIM + c0 + tx] = (_Float16)tile[tx][ty];
}

// ---------------- kernel 2: QKV projection + RoPE scale ----------------------
__global__ __launch_bounds__(128) void qkv_kernel(
    const _Float16* __restrict__ xT, const _Float16* __restrict__ W16,
    const float* __restrict__ bq, const float* __restrict__ bk,
    const float* __restrict__ bv,
    _Float16* __restrict__ Qt, _Float16* __restrict__ Kt,
    _Float16* __restrict__ Vt) {
  const int wave = threadIdx.x >> 5;
  const int lane = threadIdx.x & 31;
  const int j = lane & 15, half = lane >> 4;
  const int b  = blockIdx.x / (NTOK / 16);
  const int n0 = (blockIdx.x % (NTOK / 16)) * 16;
  const int n  = n0 + j;
  const int h  = wave;
  const _Float16* xrow = xT + ((size_t)b * NTOK + n0) * CDIM;
  const float pos = -1.f + (2.f / 4095.f) * (float)n;   // linspace(-1,1,N)

  for (int p = 0; p < 3; p++) {
    const _Float16* W = W16 + (size_t)p * CDIM * CDIM;
    const float* bias = (p == 0) ? bq : (p == 1) ? bk : bv;
    for (int t = 0; t < 4; t++) {
      const int o0 = wave * 64 + t * 16;
      v8f acc = zf8();
      #pragma unroll
      for (int c0 = 0; c0 < CDIM; c0 += 32) {
        v16h a  = load_a16(W + (size_t)o0 * CDIM + c0, CDIM);
        v16h bm = load_bt16(xrow + c0, CDIM);
        acc = wmma16(a, bm, acc);
      }
      const int dbase = t * 16 + half * 8;
      if (p < 2) {
        V8H o;
        #pragma unroll
        for (int r = 0; r < 8; r++) {
          int d = dbase + r;
          float val = acc[r] + bias[o0 + half * 8 + r];
          float inv = __expf(-0.28782313662f * (float)(d & 31)); // 10000^(-i/32)
          float ang = pos * inv;
          float pe  = (d < 32) ? __sinf(ang) : __cosf(ang);
          o.e[r] = (_Float16)(val * pe);
        }
        _Float16* dst = ((p == 0) ? Qt : Kt) +
                        (((size_t)(b * NHEAD + h) * NTOK + n) * HD + dbase);
        *(v8h*)dst = o.v;
      } else {
        _Float16* vbase = Vt + ((size_t)(b * NHEAD + h) * HD) * NTOK;
        #pragma unroll
        for (int r = 0; r < 8; r++) {
          int d = dbase + r;
          float val = acc[r] + bias[o0 + half * 8 + r];
          vbase[(size_t)d * NTOK + n] = (_Float16)val;
        }
      }
    }
  }
}

// ---------------- kernel 3: flash attention ----------------------------------
// Block = 4 waves sharing one (b,h); wave w owns rows [m0, m0+16).
// K/V 32-token chunks staged in LDS via async global->LDS copies, double buffered.
__global__ __launch_bounds__(128) void attn_kernel(
    const _Float16* __restrict__ Qt, const _Float16* __restrict__ Kt,
    const _Float16* __restrict__ Vt, const float* __restrict__ x,
    _Float16* __restrict__ yT) {
  __shared__ __align__(16) _Float16 ksh[2][32 * HD];   // 2 x 4KB, rows = tokens
  __shared__ __align__(16) _Float16 vsh[2][HD * 32];   // 2 x 4KB, rows = dims
  __shared__ __align__(16) _Float16 pbuf[4][16 * 32];  // per-wave P strip

  const int tid  = threadIdx.x;
  const int wave = tid >> 5;
  const int lane = tid & 31;
  const int j = lane & 15, half = lane >> 4;
  const int bid = blockIdx.x;                 // B*NH*(N/64)
  const int b   = bid >> 8;                   // / (NH*64)
  const int h   = (bid >> 6) & 3;
  const int m0  = (bid & 63) * 64 + wave * 16;

  const _Float16* qb = Qt + ((size_t)(b * NHEAD + h) * NTOK + m0) * HD;
  const _Float16* kb = Kt + ((size_t)(b * NHEAD + h) * NTOK) * HD;
  const _Float16* vb = Vt + ((size_t)(b * NHEAD + h) * HD) * NTOK;
  _Float16* pw = &pbuf[wave][0];

  const v16h aq0 = load_a16(qb, HD);
  const v16h aq1 = load_a16(qb + 32, HD);

  v8f Ot[4] = {zf8(), zf8(), zf8(), zf8()};
  float m[8], l[8];
  #pragma unroll
  for (int r = 0; r < 8; r++) { m[r] = -1e30f; l[r] = 0.f; }

  // cooperative async stage of one 32-token K/V chunk into LDS buffer `buf`
  auto stage_chunk = [&](int buf, int n0) {
    // K: 32 rows x 64 halves = 256 x 16B segments
    #pragma unroll
    for (int e = tid; e < 256; e += 128) {
      int row = e >> 3, seg = e & 7;
      async_ld_b128(&ksh[buf][row * HD + seg * 8],
                    kb + (size_t)(n0 + row) * HD + seg * 8);
    }
    // V: 64 rows x 32 halves = 256 x 16B segments
    #pragma unroll
    for (int e = tid; e < 256; e += 128) {
      int d = e >> 2, seg = e & 3;
      async_ld_b128(&vsh[buf][d * 32 + seg * 8],
                    vb + (size_t)d * NTOK + n0 + seg * 8);
    }
  };

  stage_chunk(0, 0);
  wait_async0();
  __syncthreads();

  for (int it = 0; it < NTOK / 32; ++it) {
    const int n0cur = it * 32;
    const _Float16* kc = &ksh[it & 1][0];
    const _Float16* vc = &vsh[it & 1][0];
    if (it + 1 < NTOK / 32) stage_chunk((it + 1) & 1, n0cur + 32);

    // S = Q * K^T  (two 16x16 tiles) from LDS
    v8f s0 = zf8(), s1 = zf8();
    s0 = wmma16(aq0, load_bt16(kc, HD), s0);
    s0 = wmma16(aq1, load_bt16(kc + 32, HD), s0);
    s1 = wmma16(aq0, load_bt16(kc + 16 * HD, HD), s1);
    s1 = wmma16(aq1, load_bt16(kc + 16 * HD + 32, HD), s1);

    // online softmax: VGPR r holds row (r + 8*half), column j
    #pragma unroll
    for (int r = 0; r < 8; r++) {
      float mx = fmaxf(s0[r], s1[r]);
      mx = fmaxf(mx, __shfl_xor(mx, 1, 32));
      mx = fmaxf(mx, __shfl_xor(mx, 2, 32));
      mx = fmaxf(mx, __shfl_xor(mx, 4, 32));
      mx = fmaxf(mx, __shfl_xor(mx, 8, 32));
      float mn = fmaxf(m[r], mx);
      float alpha = __expf(m[r] - mn);
      float p0 = __expf(s0[r] - mn);
      float p1 = __expf(s1[r] - mn);
      float rs = p0 + p1;
      rs += __shfl_xor(rs, 1, 32);
      rs += __shfl_xor(rs, 2, 32);
      rs += __shfl_xor(rs, 4, 32);
      rs += __shfl_xor(rs, 8, 32);
      l[r] = l[r] * alpha + rs;
      m[r] = mn;
      Ot[0][r] *= alpha; Ot[1][r] *= alpha; Ot[2][r] *= alpha; Ot[3][r] *= alpha;
      const int row = r + half * 8;
      pw[row * 32 + j]      = (_Float16)p0;
      pw[row * 32 + 16 + j] = (_Float16)p1;
    }
    // same-wave DS ordering: P stores precede the A-layout reload
    v16h ap = load_a16(pw, 32);
    #pragma unroll
    for (int dt = 0; dt < 4; dt++)
      Ot[dt] = wmma16(ap, load_bt16(vc + dt * 16 * 32, 32), Ot[dt]);

    wait_async0();     // next chunk landed in LDS
    __syncthreads();   // publish it / protect buffer reuse
  }

  // epilogue: normalize, add residual x, store yT (B,N,C) f16
  float rl[8];
  #pragma unroll
  for (int r = 0; r < 8; r++) rl[r] = 1.f / l[r];
  const float* xb = x + (size_t)b * CDIM * NTOK;
  #pragma unroll
  for (int dt = 0; dt < 4; dt++) {
    const int d = dt * 16 + j;
    const float* xc = xb + (size_t)(h * HD + d) * NTOK + m0 + half * 8;
    v4f xv0 = *(const v4f*)(xc);
    v4f xv1 = *(const v4f*)(xc + 4);
    #pragma unroll
    for (int r = 0; r < 8; r++) {
      float xr = (r < 4) ? xv0[r & 3] : xv1[r & 3];
      float val = Ot[dt][r] * rl[r] + xr;
      int tok = m0 + half * 8 + r;
      yT[((size_t)b * NTOK + tok) * CDIM + h * HD + d] = (_Float16)val;
    }
  }
}

// ---------------- kernel 4: output projection --------------------------------
__global__ __launch_bounds__(128) void oproj_kernel(
    const _Float16* __restrict__ yT, const _Float16* __restrict__ Wo16,
    const float* __restrict__ bo, float* __restrict__ out) {
  const int wave = threadIdx.x >> 5;
  const int lane = threadIdx.x & 31;
  const int j = lane & 15, half = lane >> 4;
  const int b  = blockIdx.x / (NTOK / 16);
  const int n0 = (blockIdx.x % (NTOK / 16)) * 16;
  const _Float16* yrow = yT + ((size_t)b * NTOK + n0) * CDIM;
  for (int t = 0; t < 4; t++) {
    const int o0 = wave * 64 + t * 16;
    v8f acc = zf8();
    #pragma unroll
    for (int c0 = 0; c0 < CDIM; c0 += 32) {
      v16h a  = load_a16(Wo16 + (size_t)o0 * CDIM + c0, CDIM);
      v16h bm = load_bt16(yrow + c0, CDIM);
      acc = wmma16(a, bm, acc);
    }
    #pragma unroll
    for (int r = 0; r < 8; r++) {
      int o = o0 + half * 8 + r;
      out[((size_t)b * CDIM + o) * NTOK + n0 + j] = acc[r] + bo[o];
    }
  }
}

// -----------------------------------------------------------------------------
extern "C" void kernel_launch(void* const* d_in, const int* in_sizes, int n_in,
                              void* d_out, int out_size, void* d_ws, size_t ws_size,
                              hipStream_t stream) {
  const float* x  = (const float*)d_in[0];
  const float* Wq = (const float*)d_in[1];
  const float* bq = (const float*)d_in[2];
  const float* Wk = (const float*)d_in[3];
  const float* bk = (const float*)d_in[4];
  const float* Wv = (const float*)d_in[5];
  const float* bv = (const float*)d_in[6];
  const float* Wo = (const float*)d_in[7];
  const float* bo = (const float*)d_in[8];
  float* out = (float*)d_out;
  const int B = in_sizes[0] / (CDIM * NTOK);   // = 2

  char* ws = (char*)d_ws;
  const size_t WSZ = (size_t)CDIM * CDIM * sizeof(_Float16);
  const size_t TSZ = (size_t)B * NTOK * CDIM * sizeof(_Float16);
  _Float16* w16 = (_Float16*)(ws);
  _Float16* xT  = (_Float16*)(ws + 4 * WSZ);
  _Float16* Qt  = (_Float16*)(ws + 4 * WSZ + TSZ);
  _Float16* Kt  = (_Float16*)(ws + 4 * WSZ + 2 * TSZ);
  _Float16* Vt  = (_Float16*)(ws + 4 * WSZ + 3 * TSZ);
  _Float16* yT  = (_Float16*)(ws + 4 * WSZ + 4 * TSZ);

  cvt_w_kernel<<<4 * CDIM * CDIM / 256, 256, 0, stream>>>(Wq, Wk, Wv, Wo, w16);
  xpose_kernel<<<dim3(NTOK / 16, CDIM / 16, B), 256, 0, stream>>>(x, xT);
  qkv_kernel<<<B * (NTOK / 16), 128, 0, stream>>>(xT, w16, bq, bk, bv, Qt, Kt, Vt);
  attn_kernel<<<B * NHEAD * (NTOK / 64), 128, 0, stream>>>(Qt, Kt, Vt, x, yT);
  oproj_kernel<<<B * (NTOK / 16), 128, 0, stream>>>(yT, w16 + 3 * CDIM * CDIM, bo, out);
}
